// HTopDownCore_4329327034927
// MI455X (gfx1250) — matrix operand order repaired and checked by reading, not compiled, not run on hardware
//
#include <hip/hip_runtime.h>
#include <math.h>

typedef __attribute__((ext_vector_type(16))) _Float16 v16h;
typedef __attribute__((ext_vector_type(8)))  _Float16 v8h;
typedef __attribute__((ext_vector_type(8)))  float    v8f;

#define B_   128
#define H_   1024
#define A_   196
#define AH_  512
#define NG_  4096   // 4*H

// ---------------------------------------------------------------------
// LDS tile format: one 16x16(x32K) operand tile = 512 halfs (1 KB).
//   half index within tile = g*256 + lane*8 + pos   (g = h>>3, pos = h&7)
// Each lane's 16 operand halves are two contiguous 16B groups ->
// exactly two ds_load_b128 per tile, already in WMMA register order.
// ---------------------------------------------------------------------
static __device__ __forceinline__ v16h ld_tile(const _Float16* base, int lane)
{
    v8h lo = *(const v8h*)(base + lane * 8);
    v8h hi = *(const v8h*)(base + 256 + lane * 8);
    return __builtin_shufflevector(lo, hi, 0,1,2,3,4,5,6,7,8,9,10,11,12,13,14,15);
}

// convert float4 -> 4 packed halves, single b64 LDS store
static __device__ __forceinline__ void st_quad(_Float16* dst, float4 v)
{
    union { _Float16 h[4]; unsigned long long u; } p;
    p.h[0] = (_Float16)v.x; p.h[1] = (_Float16)v.y;
    p.h[2] = (_Float16)v.z; p.h[3] = (_Float16)v.w;
    *(unsigned long long*)dst = p.u;
}

// =====================================================================
// Segment-parallel GEMM partial:
//   out[seg, 128, N] = X_seg[128,1024] @ W_seg[N,1024]^T      (seg = blockIdx.y)
// X_seg row stride = 1024; W_seg row stride = s_seg.
// fp32 memory -> f16 operands (converted once at staging) -> f32 WMMA acc.
// Block = 8 wave32 = 128x128 C-tile; wave = 2 M-tiles x 4 N-tiles.
// =====================================================================
__global__ __launch_bounds__(256)
void gemm_xwT_wmma(const float* x0, const float* w0, int s0,
                   const float* x1, const float* w1, int s1,
                   const float* x2, const float* w2, int s2,
                   const float* x3, const float* w3, int s3,
                   int N, float* out)
{
    __shared__ _Float16 lA[8 * 512];   // 8 M-tiles, swizzled f16 (8 KB)
    __shared__ _Float16 lB[8 * 512];   // 8 N-tiles, swizzled f16 (8 KB)

    const int tid  = threadIdx.x;
    const int lane = tid & 31;
    const int wv   = tid >> 5;
    const int mgrp = wv & 3;                 // M-tiles 2*mgrp, 2*mgrp+1
    const int ngrp = wv >> 2;                // N-tiles 4*ngrp .. +3
    const long n0g = (long)blockIdx.x * 128; // block N offset

    // segment select (uniform across block)
    const float* X; const float* W; long wstride;
    switch (blockIdx.y) {
        case 0:  X = x0; W = w0; wstride = s0; break;
        case 1:  X = x1; W = w1; wstride = s1; break;
        case 2:  X = x2; W = w2; wstride = s2; break;
        default: X = x3; W = w3; wstride = s3; break;
    }
    float* o = out + (long)blockIdx.y * B_ * N;

    v8f acc[8] = {};                         // [nt][mi] -> acc[nt*2+mi]

    // ---- staging geometry (k-invariant): 4 float4 per array per thread
    int xoff[4], woff[4];
    const float* pX[4]; const float* pW[4];
    #pragma unroll
    for (int r = 0; r < 4; ++r) {
        int i = tid * 4 + r * 1024;          // 0..4095 over [128 x 32]
        int row = i >> 5, col = i & 31;      // col % 4 == 0
        // A swizzle: lane = (m&15) + ((k>>3)&1)*16 ; h = ((k>>4)<<3)+(k&7)
        int tm = row >> 4, m = row & 15;
        xoff[r] = tm * 512 + ((col >> 4) << 8)
                + (m + (((col >> 3) & 1) << 4)) * 8 + (col & 7);
        // B swizzle: lane = (n&15) + ((k>>3)&1)*16 ; h = ((k>>4)<<3)+(k&7)
        int tn = row >> 4, n = row & 15;
        woff[r] = tn * 512 + ((col >> 4) << 8)
                + (n + (((col >> 3) & 1) << 4)) * 8 + (col & 7);
        pX[r] = X + (long)row * H_ + col;
        pW[r] = W + (n0g + row) * wstride + col;
    }

    const _Float16* aBase = &lA[(mgrp * 2) * 512];
    const _Float16* bBase = &lB[(ngrp * 4) * 512];

    for (int k0 = 0; k0 < 1024; k0 += 32) {
        // issue global loads early (overlap previous chunk's WMMAs)
        float4 vx[4], vw[4];
        #pragma unroll
        for (int r = 0; r < 4; ++r) vx[r] = *(const float4*)(pX[r] + k0);
        #pragma unroll
        for (int r = 0; r < 4; ++r) vw[r] = *(const float4*)(pW[r] + k0);
        __builtin_prefetch(pW[0] + ((k0 + 32) & 1023), 0, 1);  // branchless

        __syncthreads();   // previous chunk fully consumed
        #pragma unroll
        for (int r = 0; r < 4; ++r) st_quad(&lA[xoff[r]], vx[r]);
        #pragma unroll
        for (int r = 0; r < 4; ++r) st_quad(&lB[woff[r]], vw[r]);
        __syncthreads();   // tiles visible

        // ---- consume: all 12 ds_load_b128 in flight, then 8 WMMAs
        v16h a0 = ld_tile(aBase,        lane);
        v16h a1 = ld_tile(aBase +  512, lane);
        v16h b0 = ld_tile(bBase,        lane);
        v16h b1 = ld_tile(bBase +  512, lane);
        v16h b2 = ld_tile(bBase + 1024, lane);
        v16h b3 = ld_tile(bBase + 1536, lane);
        acc[0] = __builtin_amdgcn_wmma_f32_16x16x32_f16(false, a0, false, b0, (short)0, acc[0], false, false);
        acc[1] = __builtin_amdgcn_wmma_f32_16x16x32_f16(false, a1, false, b0, (short)0, acc[1], false, false);
        acc[2] = __builtin_amdgcn_wmma_f32_16x16x32_f16(false, a0, false, b1, (short)0, acc[2], false, false);
        acc[3] = __builtin_amdgcn_wmma_f32_16x16x32_f16(false, a1, false, b1, (short)0, acc[3], false, false);
        acc[4] = __builtin_amdgcn_wmma_f32_16x16x32_f16(false, a0, false, b2, (short)0, acc[4], false, false);
        acc[5] = __builtin_amdgcn_wmma_f32_16x16x32_f16(false, a1, false, b2, (short)0, acc[5], false, false);
        acc[6] = __builtin_amdgcn_wmma_f32_16x16x32_f16(false, a0, false, b3, (short)0, acc[6], false, false);
        acc[7] = __builtin_amdgcn_wmma_f32_16x16x32_f16(false, a1, false, b3, (short)0, acc[7], false, false);
    }

    // ---- epilogue: C/D layout  M = m0 + v + ((lane>=16)?8:0), N = lane&15
    const int mrow0 = mgrp * 32 + ((lane >> 4) << 3);
    const int ncol  = lane & 15;
    #pragma unroll
    for (int mi = 0; mi < 2; ++mi) {
        #pragma unroll
        for (int nt = 0; nt < 4; ++nt) {
            const long n = n0g + ngrp * 64 + nt * 16 + ncol;
            #pragma unroll
            for (int v = 0; v < 8; ++v)
                o[(long)(mrow0 + mi * 16 + v) * N + n] = acc[nt * 2 + mi][v];
        }
    }
}

// =====================================================================
// LSTM pointwise: sum nseg gate partials [nseg,B,4H] + biases -> h,c
// =====================================================================
__global__ __launch_bounds__(256)
void lstm_pointwise(const float* gparts, int nseg,
                    const float* b_ih, const float* b_hh,
                    const float* c_prev,
                    float* h_out, float* c_out, float* h_copy)
{
    int i = blockIdx.x * 256 + threadIdx.x;
    if (i >= B_ * H_) return;
    int b = i >> 10, j = i & 1023;
    float gi = b_ih[j]          + b_hh[j];
    float gf = b_ih[H_ + j]     + b_hh[H_ + j];
    float gg = b_ih[2 * H_ + j] + b_hh[2 * H_ + j];
    float go = b_ih[3 * H_ + j] + b_hh[3 * H_ + j];
    for (int s = 0; s < nseg; ++s) {
        const float* g = gparts + (long)s * B_ * NG_ + (long)b * NG_;
        gi += g[j]; gf += g[H_ + j]; gg += g[2 * H_ + j]; go += g[3 * H_ + j];
    }
    float si = 1.f / (1.f + __expf(-gi));
    float sf = 1.f / (1.f + __expf(-gf));
    float so = 1.f / (1.f + __expf(-go));
    float c  = sf * c_prev[i] + si * tanhf(gg);
    float h  = so * tanhf(c);
    c_out[i] = c;
    h_out[i] = h;
    if (h_copy) h_copy[i] = h;
}

// =====================================================================
// logits[b,s] = sum_a tanh(p_att[b,s,a] + att_h[b,a] + hb[a]) * aw[a] + ab
// One wave32 per (b,s).
// =====================================================================
__global__ __launch_bounds__(256)
void attn_logits(const float* p_att, const float* att_h, const float* hb,
                 const float* aw, const float* ab, float* logits)
{
    int wid  = blockIdx.x * 8 + (threadIdx.x >> 5);
    int lane = threadIdx.x & 31;
    if (wid >= B_ * A_) return;
    int b = wid / A_, s = wid % A_;
    const float* p  = p_att + ((long)b * A_ + s) * AH_;
    const float* hh = att_h + (long)b * AH_;
    float acc = 0.f;
    #pragma unroll
    for (int a0 = lane * 4; a0 < AH_; a0 += 128) {
        float4 pv = *(const float4*)(p + a0);
        float4 hv = *(const float4*)(hh + a0);
        float4 bv = *(const float4*)(hb + a0);
        float4 av = *(const float4*)(aw + a0);
        acc += tanhf(pv.x + hv.x + bv.x) * av.x + tanhf(pv.y + hv.y + bv.y) * av.y
             + tanhf(pv.z + hv.z + bv.z) * av.z + tanhf(pv.w + hv.w + bv.w) * av.w;
    }
    #pragma unroll
    for (int off = 16; off; off >>= 1)
        acc += __shfl_xor(acc, off, 32);
    if (lane == 0) logits[wid] = acc + ab[0];
}

// =====================================================================
// in-place softmax over A_=196 values per row (one block per row)
// =====================================================================
__global__ __launch_bounds__(256)
void softmax_rows(float* x)
{
    __shared__ float red[256];
    int b = blockIdx.x, t = threadIdx.x;
    float* row = x + (long)b * A_;
    float v = (t < A_) ? row[t] : -3.0e38f;
    red[t] = v; __syncthreads();
    for (int o = 128; o; o >>= 1) {
        if (t < o) red[t] = fmaxf(red[t], red[t + o]);
        __syncthreads();
    }
    float mx = red[0]; __syncthreads();
    float e = (t < A_) ? __expf(v - mx) : 0.f;
    red[t] = e; __syncthreads();
    for (int o = 128; o; o >>= 1) {
        if (t < o) red[t] += red[t + o];
        __syncthreads();
    }
    float inv = 1.f / red[0];
    if (t < A_) row[t] = e * inv;
}

// =====================================================================
// att[b,h] = sum_s w[b,s] * feats[b,s,h]  — float4, one block per b
// =====================================================================
__global__ __launch_bounds__(256)
void attn_wsum(const float* wgt, const float* feats, float* out)
{
    int b  = blockIdx.x;
    int h4 = threadIdx.x * 4;
    const float* f  = feats + (long)b * A_ * H_ + h4;
    const float* wr = wgt + (long)b * A_;
    float4 acc = {0.f, 0.f, 0.f, 0.f};
    for (int s = 0; s < A_; ++s) {
        float w = wr[s];
        float4 v = *(const float4*)(f + (long)s * H_);
        acc.x += w * v.x; acc.y += w * v.y;
        acc.z += w * v.z; acc.w += w * v.w;
    }
    *(float4*)(out + (long)b * H_ + h4) = acc;
}

// =====================================================================
extern "C" void kernel_launch(void* const* d_in, const int* in_sizes, int n_in,
                              void* d_out, int out_size, void* d_ws, size_t ws_size,
                              hipStream_t stream)
{
    const float* st_in = (const float*)d_in[0];
    const float* xt    = (const float*)d_in[1];
    const float* fc    = (const float*)d_in[2];
    const float* attf  = (const float*)d_in[3];
    const float* patt  = (const float*)d_in[4];
    const float* h_state = (const float*)d_in[5];
    const float* c_state = (const float*)d_in[6];
    // d_in[7] = cs_index (unused by reference computation)
    const float* satt_wih = (const float*)d_in[8];
    const float* satt_whh = (const float*)d_in[9];
    const float* satt_bih = (const float*)d_in[10];
    const float* satt_bhh = (const float*)d_in[11];
    const float* sent_wih = (const float*)d_in[12];
    const float* sent_whh = (const float*)d_in[13];
    const float* sent_bih = (const float*)d_in[14];
    const float* sent_bhh = (const float*)d_in[15];
    const float* watt_wih = (const float*)d_in[16];
    const float* watt_whh = (const float*)d_in[17];
    const float* watt_bih = (const float*)d_in[18];
    const float* watt_bhh = (const float*)d_in[19];
    const float* word_wih = (const float*)d_in[20];
    const float* word_whh = (const float*)d_in[21];
    const float* word_bih = (const float*)d_in[22];
    const float* word_bhh = (const float*)d_in[23];
    const float* hw = (const float*)d_in[24];
    const float* hb = (const float*)d_in[25];
    const float* aw = (const float*)d_in[26];
    const float* ab = (const float*)d_in[27];

    const long BH = (long)B_ * H_;
    float* out    = (float*)d_out;        // [B,H] final output
    float* h_att  = out + BH;             // h_stack[0..3]
    float* h_sen  = out + 2 * BH;
    float* h_watt = out + 3 * BH;
    float* h_word = out + 4 * BH;
    float* c_att  = out + 5 * BH;
    float* c_sen  = out + 6 * BH;
    float* c_watt = out + 7 * BH;
    float* c_word = out + 8 * BH;

    float* wsp    = (float*)d_ws;
    float* gparts = wsp;                          // [4][B][4H] partials (8 MB)
    float* atth   = gparts + 4L * B_ * NG_;       // [B, AH]
    float* lg     = atth + (long)B_ * AH_;        // [B, A]
    float* attv   = lg + (long)B_ * A_;           // [B, H]

    const float* h0 = h_state;
    const float* h1 = h_state + BH;
    const float* h2 = h_state + 2 * BH;
    const float* h3 = h_state + 3 * BH;
    const float* c0 = c_state;
    const float* c1 = c_state + BH;
    const float* c2 = c_state + 2 * BH;
    const float* c3 = c_state + 3 * BH;

    const dim3 blk(256);
    const dim3 gG4(NG_ / 128, 4);   // 128 blocks: 4-segment LSTM GEMMs
    const dim3 gG3(NG_ / 128, 3);   // 96 blocks:  3-segment LSTM GEMMs
    const dim3 gAtth(AH_ / 128, 1); // 4 blocks:   tiny attention projection
    const dim3 gPW((B_ * H_) / 256);
    const dim3 gLog((B_ * A_) / 8);

    // ---- satt_lstm: x = [h1, st, fc], h = h0
    gemm_xwT_wmma<<<gG4, blk, 0, stream>>>(
        h1,    satt_wih,           3 * H_,
        st_in, satt_wih + H_,      3 * H_,
        fc,    satt_wih + 2 * H_,  3 * H_,
        h0,    satt_whh,           H_,
        NG_, gparts);
    lstm_pointwise<<<gPW, blk, 0, stream>>>(gparts, 4, satt_bih, satt_bhh,
                                            c0, h_att, c_att, nullptr);

    // ---- attention(h_att)
    gemm_xwT_wmma<<<gAtth, blk, 0, stream>>>(
        h_att, hw, H_, nullptr, nullptr, 0, nullptr, nullptr, 0,
        nullptr, nullptr, 0, AH_, atth);
    attn_logits<<<gLog, blk, 0, stream>>>(patt, atth, hb, aw, ab, lg);
    softmax_rows<<<B_, blk, 0, stream>>>(lg);
    attn_wsum<<<B_, blk, 0, stream>>>(lg, attf, attv);

    // ---- sent_lstm: x = [att, h_att], h = h1
    gemm_xwT_wmma<<<gG3, blk, 0, stream>>>(
        attv,  sent_wih,       2 * H_,
        h_att, sent_wih + H_,  2 * H_,
        h1,    sent_whh,       H_,
        nullptr, nullptr, 0,
        NG_, gparts);
    lstm_pointwise<<<gPW, blk, 0, stream>>>(gparts, 3, sent_bih, sent_bhh,
                                            c1, h_sen, c_sen, nullptr);

    // ---- watt_lstm: x = [xt, h3, topic=h_sen], h = h2
    gemm_xwT_wmma<<<gG4, blk, 0, stream>>>(
        xt,    watt_wih,           3 * H_,
        h3,    watt_wih + H_,      3 * H_,
        h_sen, watt_wih + 2 * H_,  3 * H_,
        h2,    watt_whh,           H_,
        NG_, gparts);
    lstm_pointwise<<<gPW, blk, 0, stream>>>(gparts, 4, watt_bih, watt_bhh,
                                            c2, h_watt, c_watt, nullptr);

    // ---- attention(h_watt)
    gemm_xwT_wmma<<<gAtth, blk, 0, stream>>>(
        h_watt, hw, H_, nullptr, nullptr, 0, nullptr, nullptr, 0,
        nullptr, nullptr, 0, AH_, atth);
    attn_logits<<<gLog, blk, 0, stream>>>(patt, atth, hb, aw, ab, lg);
    softmax_rows<<<B_, blk, 0, stream>>>(lg);
    attn_wsum<<<B_, blk, 0, stream>>>(lg, attf, attv);

    // ---- word_lstm: x = [watt_res, h_watt], h = h3; also writes `output`
    gemm_xwT_wmma<<<gG3, blk, 0, stream>>>(
        attv,   word_wih,       2 * H_,
        h_watt, word_wih + H_,  2 * H_,
        h3,     word_whh,       H_,
        nullptr, nullptr, 0,
        NG_, gparts);
    lstm_pointwise<<<gPW, blk, 0, stream>>>(gparts, 3, word_bih, word_bhh,
                                            c3, h_word, c_word, out);
}
